// Compressive_Representation_55542517072464
// MI455X (gfx1250) — matrix operand order repaired and checked by reading, not compiled, not run on hardware
//
#include <hip/hip_runtime.h>
#include <stdint.h>

// ---------------- problem constants ----------------
#define NPATCH 7056
#define NPS    84
#define CCH    4
#define TS     7
#define HW     49
#define TSR    28
#define HWR    784
#define STR    3
#define STRR   12
#define SLR    256
#define SHR    1024
#define NAT    10
#define KDIM   196      // C*TS*TS
#define KP     224      // padded to multiple of 32 for wmma bf16 k=32
#define RBLK   441      // 7056/16 row blocks
#define CTILES 441      // 7056/16 col tiles
#define COLCHUNK 56     // ceil(441/8 waves)

typedef __attribute__((ext_vector_type(16))) __bf16 v16bf;
typedef __attribute__((ext_vector_type(8)))  float  v8f;
typedef __attribute__((ext_vector_type(4)))  float  f32x4;

__device__ __forceinline__ __bf16 f2bf(float f) {
  unsigned u = __float_as_uint(f);
  u += 0x7FFFu + ((u >> 16) & 1u);             // RNE
  unsigned short s = (unsigned short)(u >> 16);
  __bf16 b; __builtin_memcpy(&b, &s, 2); return b;
}

__device__ __forceinline__ unsigned long long shflxor64(unsigned long long v, int m) {
  int lo = __shfl_xor((int)(unsigned)(v & 0xffffffffull), m, 32);
  int hi = __shfl_xor((int)(unsigned)(v >> 32), m, 32);
  return ((unsigned long long)(unsigned)hi << 32) | (unsigned)lo;
}

// ---------------- 1) wald blur (separable 17-tap gaussian, sigma=4), sampled at 2+4j ----------------
__global__ __launch_bounds__(256) void k_blur_v(const float* __restrict__ hrms, float* __restrict__ tmp) {
  int t = blockIdx.x * blockDim.x + threadIdx.x;
  if (t >= CCH * SLR * SHR) return;
  int x = t % SHR; int j = (t / SHR) % SLR; int c = t / (SHR * SLR);
  float w[17], ws = 0.f;
  #pragma unroll
  for (int k = 0; k < 17; ++k) { float d = (float)(k - 8); w[k] = __expf(-d * d * (1.f/32.f)); ws += w[k]; }
  float inv = 1.f / ws;
  int y0 = 2 + 4 * j - 8;
  const float* p = hrms + (size_t)c * SHR * SHR + x;
  float acc = 0.f;
  #pragma unroll
  for (int k = 0; k < 17; ++k) {
    int y = y0 + k;
    if (y >= 0 && y < SHR) acc += w[k] * p[(size_t)y * SHR];
  }
  tmp[t] = acc * inv;
}

__global__ __launch_bounds__(256) void k_blur_h_diff(const float* __restrict__ tmp, const float* __restrict__ ms_hp,
                                                     float* __restrict__ diff) {
  int t = blockIdx.x * blockDim.x + threadIdx.x;
  if (t >= CCH * SLR * SLR) return;
  int i = t % SLR; int j = (t / SLR) % SLR; int c = t / (SLR * SLR);
  float w[17], ws = 0.f;
  #pragma unroll
  for (int k = 0; k < 17; ++k) { float d = (float)(k - 8); w[k] = __expf(-d * d * (1.f/32.f)); ws += w[k]; }
  float inv = 1.f / ws;
  int x0 = 2 + 4 * i - 8;
  const float* p = tmp + ((size_t)c * SLR + j) * SHR;
  float acc = 0.f;
  #pragma unroll
  for (int k = 0; k < 17; ++k) {
    int x = x0 + k;
    if (x >= 0 && x < SHR) acc += w[k] * p[x];
  }
  diff[t] = ms_hp[t] - acc * inv;
}

// ---------------- 2) unfold residual + bf16 pack; init idx col0 ----------------
__global__ __launch_bounds__(256) void k_unfold(const float* __restrict__ diff, float* __restrict__ res_ms,
                                                __bf16* __restrict__ resb, int* __restrict__ idx) {
  int b = blockIdx.x * blockDim.x + threadIdx.x;
  if (b >= NPATCH * CCH) return;
  int n = b >> 2, c = b & 3;
  int py = n / NPS, px = n % NPS;
  const float* src = diff + ((size_t)c * SLR + py * STR) * SLR + px * STR;
  float* dst = res_ms + (size_t)b * HW;
  __bf16* dstb = resb + (size_t)n * KP + c * HW;
  for (int dy = 0; dy < TS; ++dy)
    for (int dx = 0; dx < TS; ++dx) {
      float v = src[(size_t)dy * SLR + dx];
      dst[dy * TS + dx] = v;
      dstb[dy * TS + dx] = f2bf(v);
    }
  if (c == 0) {
    idx[n * NAT] = n;                               // index[:,0] = arange
    for (int k = KDIM; k < KP; ++k) resb[(size_t)n * KP + k] = f2bf(0.f);
  }
}

__global__ __launch_bounds__(256) void k_dictcvt(const float* __restrict__ lr_dict, __bf16* __restrict__ dictb) {
  int t = blockIdx.x * blockDim.x + threadIdx.x;
  if (t >= NPATCH * KP) return;
  int a = t / KP, k = t % KP;
  dictb[t] = (k < KDIM) ? f2bf(lr_dict[(size_t)a * KDIM + k]) : f2bf(0.f);
}

__global__ __launch_bounds__(256) void k_zero_best(unsigned long long* best) {
  int t = blockIdx.x * blockDim.x + threadIdx.x;
  if (t < NPATCH) best[t] = 0ull;
}

// ---------------- 3) WMMA correlation GEMM fused with per-row abs-argmax ----------------
// scores[n,m] = res[n,:196] . lr_dict[m,:196]; keep argmax_m |score| per row n.
// Wave = one 16-row block x COLCHUNK column tiles; bf16 16x16x32, K padded to 224 (7 steps).
__global__ __launch_bounds__(256) void k_corr(const __bf16* __restrict__ resb, const __bf16* __restrict__ dictb,
                                              unsigned long long* __restrict__ best) {
  int rb   = blockIdx.x;                    // 0..440
  int wave = threadIdx.x >> 5;              // 0..7 -> column chunk
  int lane = threadIdx.x & 31;
  int half = lane >> 4;
  int lq   = lane & 15;
  int rowbase = rb * 16;
  unsigned long long bk[8];
  #pragma unroll
  for (int r = 0; r < 8; ++r) bk[r] = 0ull;

  int ct0 = wave * COLCHUNK;
  int ct1 = ct0 + COLCHUNK; if (ct1 > CTILES) ct1 = CTILES;

  const __bf16* arow = resb + (size_t)(rowbase + lq) * KP;   // A: M = rowbase+lq
  for (int ct = ct0; ct < ct1; ++ct) {
    int colbase = ct * 16;
    const __bf16* brow = dictb + (size_t)(colbase + lq) * KP; // B column n = colbase+lq
    v8f acc = {0.f,0.f,0.f,0.f,0.f,0.f,0.f,0.f};
    #pragma unroll
    for (int ks = 0; ks < KP; ks += 32) {
      // A 16x32 bf16 layout: lane(half,m): elems0-7 -> K=ks+8*half+0..7, elems8-15 -> K=ks+16+8*half+0..7
      union { v16bf v; f32x4 f[2]; } A;
      A.f[0] = *(const f32x4*)(arow + ks + 8 * half);
      A.f[1] = *(const f32x4*)(arow + ks + 16 + 8 * half);
      // B 32x16 bf16 layout: lane(half,n): elems0-7 -> K=ks+8*half+0..7, elems8-15 -> K=ks+16+8*half+0..7
      union { v16bf v; f32x4 f[2]; } B;
      B.f[0] = *(const f32x4*)(brow + ks + 8 * half);
      B.f[1] = *(const f32x4*)(brow + ks + 16 + 8 * half);
      acc = __builtin_amdgcn_wmma_f32_16x16x32_bf16(false, A.v, false, B.v, (short)0, acc, false, false);
    }
    // D layout: lane L (col n=L%16), vgpr r -> row m = 8*(L/16)+r
    #pragma unroll
    for (int r = 0; r < 8; ++r) {
      float av = __builtin_fabsf(acc[r]);
      unsigned long long key = (((unsigned long long)__float_as_uint(av)) << 32) | (unsigned)(colbase + lq);
      if (key > bk[r]) bk[r] = key;
    }
  }
  // reduce across the 16 lanes sharing a row, then one atomic per row
  #pragma unroll
  for (int r = 0; r < 8; ++r) {
    unsigned long long k = bk[r];
    #pragma unroll
    for (int m = 8; m >= 1; m >>= 1) {
      unsigned long long o = shflxor64(k, m);
      if (o > k) k = o;
    }
    if (lq == 0) atomicMax(best + rowbase + 8 * half + r, k);
  }
}

__global__ __launch_bounds__(256) void k_extract(const unsigned long long* __restrict__ best,
                                                 int* __restrict__ idx, int col) {
  int n = blockIdx.x * blockDim.x + threadIdx.x;
  if (n < NPATCH) idx[n * NAT + col] = (int)(unsigned)(best[n] & 0xffffffffull);
}

// ---------------- 4) batched tiny least-squares via normal equations (unrolled per K) ----------------
template<int K>
__global__ __launch_bounds__(128) void k_solve(const float* __restrict__ res_ms, const float* __restrict__ lr_dict,
                                               const int* __restrict__ idx, float* __restrict__ alpha,
                                               __bf16* __restrict__ resb, int writeRes) {
  int b = blockIdx.x * blockDim.x + threadIdx.x;
  if (b >= NPATCH * CCH) return;
  int n = b >> 2, c = b & 3;
  const float* cols[K];
  #pragma unroll
  for (int j = 0; j < K; ++j) cols[j] = lr_dict + (size_t)idx[n * NAT + j] * KDIM + c * HW;
  const float* y = res_ms + (size_t)b * HW;

  float G[K][K], r[K];
  #pragma unroll
  for (int j = 0; j < K; ++j) {
    r[j] = 0.f;
    #pragma unroll
    for (int l = 0; l < K; ++l) G[j][l] = 0.f;
  }
  for (int h = 0; h < HW; ++h) {
    float a[K];
    #pragma unroll
    for (int j = 0; j < K; ++j) a[j] = cols[j][h];
    float yh = y[h];
    #pragma unroll
    for (int j = 0; j < K; ++j) {
      r[j] += a[j] * yh;
      #pragma unroll
      for (int l = 0; l <= j; ++l) G[j][l] += a[j] * a[l];
    }
  }
  float tr = 0.f;
  #pragma unroll
  for (int j = 0; j < K; ++j) tr += G[j][j];
  float eps = 1e-6f * tr + 1e-12f;                 // ridge ~ pinv robustness for repeated atoms
  #pragma unroll
  for (int j = 0; j < K; ++j) {
    G[j][j] += eps;
    #pragma unroll
    for (int l = j + 1; l < K; ++l) G[j][l] = G[l][j];
  }
  #pragma unroll
  for (int p = 0; p < K; ++p) {                    // Gaussian elimination (SPD, no pivot)
    float inv = 1.f / G[p][p];
    #pragma unroll
    for (int j = p + 1; j < K; ++j) {
      float f = G[j][p] * inv;
      #pragma unroll
      for (int l = p; l < K; ++l) G[j][l] -= f * G[p][l];
      r[j] -= f * r[p];
    }
  }
  float x[K];
  #pragma unroll
  for (int p = K - 1; p >= 0; --p) {
    float s = r[p];
    #pragma unroll
    for (int l = p + 1; l < K; ++l) s -= G[p][l] * x[l];
    x[p] = s / G[p][p];
  }
  #pragma unroll
  for (int j = 0; j < K; ++j) alpha[(size_t)b * NAT + j] = x[j];

  if (writeRes) {                                  // res = y - A x  -> bf16 row for next corr GEMM
    __bf16* dstb = resb + (size_t)n * KP + c * HW;
    for (int h = 0; h < HW; ++h) {
      float v = y[h];
      #pragma unroll
      for (int j = 0; j < K; ++j) v -= x[j] * cols[j][h];
      dstb[h] = f2bf(v);
    }
  }
}

// ---------------- 5) HR reconstruction: hr_p[b,:] = sum_i alpha[b,i] * hr_dict[idx[n,i], c*784:...] ----------------
__global__ __launch_bounds__(256) void k_recon(const float* __restrict__ hr_dict, const float* __restrict__ alpha,
                                               const int* __restrict__ idx, float* __restrict__ hr_p) {
  int b = blockIdx.x;                              // 0..28223
  int n = b >> 2, c = b & 3;
  __shared__ float sa[NAT];
  __shared__ int   si[NAT];
  if (threadIdx.x < NAT) {
    sa[threadIdx.x] = alpha[(size_t)b * NAT + threadIdx.x];
    si[threadIdx.x] = idx[n * NAT + threadIdx.x];
  }
  __syncthreads();
  for (int k = threadIdx.x; k < HWR; k += 256) {
    float acc = 0.f;
    #pragma unroll
    for (int i = 0; i < NAT; ++i)
      acc += sa[i] * hr_dict[(size_t)si[i] * (CCH * HWR) + (size_t)c * HWR + k];
    hr_p[(size_t)b * HWR + k] = acc;
  }
}

// ---------------- 6) fold (overlap-add gather), divide, add hrms_hp + lms ----------------
__global__ __launch_bounds__(256) void k_fold(const float* __restrict__ hr_p, const float* __restrict__ divisor,
                                              const float* __restrict__ hrms_hp, const float* __restrict__ lms,
                                              float* __restrict__ out) {
  int t = blockIdx.x * blockDim.x + threadIdx.x;
  if (t >= CCH * SHR * SHR) return;
  int x = t & (SHR - 1);
  int y = (t >> 10) & (SHR - 1);
  int c = t >> 20;
  int py0 = (y >= TSR) ? ((y - TSR) / STRR + 1) : 0;
  int py1 = y / STRR; if (py1 > NPS - 1) py1 = NPS - 1;
  int px0 = (x >= TSR) ? ((x - TSR) / STRR + 1) : 0;
  int px1 = x / STRR; if (px1 > NPS - 1) px1 = NPS - 1;
  float acc = 0.f;
  for (int py = py0; py <= py1; ++py) {
    int dy = y - py * STRR;
    for (int px = px0; px <= px1; ++px) {
      int dx = x - px * STRR;
      int n = py * NPS + px;
      acc += hr_p[(size_t)(n * CCH + c) * HWR + dy * TSR + dx];
    }
  }
  out[t] = acc / (divisor[t] + 1e-8f) + hrms_hp[t] + lms[t];
}

// ---------------- host launcher ----------------
static inline size_t alignUp(size_t v, size_t a) { return (v + a - 1) / a * a; }

extern "C" void kernel_launch(void* const* d_in, const int* in_sizes, int n_in,
                              void* d_out, int out_size, void* d_ws, size_t ws_size,
                              hipStream_t stream) {
  const float* hrms_hp = (const float*)d_in[0];
  const float* ms_hp   = (const float*)d_in[1];
  const float* lms     = (const float*)d_in[2];
  const float* lr_dict = (const float*)d_in[3];
  const float* hr_dict = (const float*)d_in[4];
  const float* divisor = (const float*)d_in[5];
  float* out = (float*)d_out;
  (void)in_sizes; (void)n_in; (void)out_size; (void)ws_size;

  char* base = (char*)d_ws;
  size_t off = 0;
  auto carve = [&](size_t bytes) -> char* { char* q = base + off; off = alignUp(off + bytes, 256); return q; };
  float* tmp    = (float*)carve(sizeof(float) * CCH * SLR * SHR);            // 4 MB
  float* diff   = (float*)carve(sizeof(float) * CCH * SLR * SLR);            // 1 MB
  float* res_ms = (float*)carve(sizeof(float) * (size_t)NPATCH * CCH * HW);  // 5.5 MB
  __bf16* resb  = (__bf16*)carve(2ull * NPATCH * KP);                        // 3.2 MB
  __bf16* dictb = (__bf16*)carve(2ull * NPATCH * KP);                        // 3.2 MB
  int* idx      = (int*)carve(sizeof(int) * NPATCH * NAT);                   // 0.28 MB
  float* alpha  = (float*)carve(sizeof(float) * (size_t)NPATCH * CCH * NAT); // 1.1 MB
  unsigned long long* best = (unsigned long long*)carve(8ull * NPATCH);      // 56 KB
  float* hr_p   = (float*)carve(sizeof(float) * (size_t)NPATCH * CCH * HWR); // 88.5 MB

  k_blur_v     <<<(CCH * SLR * SHR + 255) / 256, 256, 0, stream>>>(hrms_hp, tmp);
  k_blur_h_diff<<<(CCH * SLR * SLR + 255) / 256, 256, 0, stream>>>(tmp, ms_hp, diff);
  k_dictcvt    <<<(NPATCH * KP + 255) / 256, 256, 0, stream>>>(lr_dict, dictb);
  k_unfold     <<<(NPATCH * CCH + 255) / 256, 256, 0, stream>>>(diff, res_ms, resb, idx);

  auto solve = [&](int i) {
    int wr = (i < NAT - 1) ? 1 : 0;
    dim3 g((NPATCH * CCH + 127) / 128), blk(128);
    switch (i + 1) {
      case 1:  k_solve<1> <<<g, blk, 0, stream>>>(res_ms, lr_dict, idx, alpha, resb, wr); break;
      case 2:  k_solve<2> <<<g, blk, 0, stream>>>(res_ms, lr_dict, idx, alpha, resb, wr); break;
      case 3:  k_solve<3> <<<g, blk, 0, stream>>>(res_ms, lr_dict, idx, alpha, resb, wr); break;
      case 4:  k_solve<4> <<<g, blk, 0, stream>>>(res_ms, lr_dict, idx, alpha, resb, wr); break;
      case 5:  k_solve<5> <<<g, blk, 0, stream>>>(res_ms, lr_dict, idx, alpha, resb, wr); break;
      case 6:  k_solve<6> <<<g, blk, 0, stream>>>(res_ms, lr_dict, idx, alpha, resb, wr); break;
      case 7:  k_solve<7> <<<g, blk, 0, stream>>>(res_ms, lr_dict, idx, alpha, resb, wr); break;
      case 8:  k_solve<8> <<<g, blk, 0, stream>>>(res_ms, lr_dict, idx, alpha, resb, wr); break;
      case 9:  k_solve<9> <<<g, blk, 0, stream>>>(res_ms, lr_dict, idx, alpha, resb, wr); break;
      default: k_solve<10><<<g, blk, 0, stream>>>(res_ms, lr_dict, idx, alpha, resb, wr); break;
    }
  };

  solve(0);
  for (int i = 1; i < NAT; ++i) {
    k_zero_best<<<(NPATCH + 255) / 256, 256, 0, stream>>>(best);
    k_corr     <<<RBLK, 256, 0, stream>>>(resb, dictb, best);
    k_extract  <<<(NPATCH + 255) / 256, 256, 0, stream>>>(best, idx, i);
    solve(i);
  }

  k_recon<<<NPATCH * CCH, 256, 0, stream>>>(hr_dict, alpha, idx, hr_p);
  k_fold <<<(CCH * SHR * SHR + 255) / 256, 256, 0, stream>>>(hr_p, divisor, hrms_hp, lms, out);
}
